// ReversibleBlock_33998961115893
// MI455X (gfx1250) — compile-verified, matrix operand order checked by reading, not run on hardware
//
#include <hip/hip_runtime.h>

// ---------------------------------------------------------------------------
// Types for CDNA5 WMMA (wave32)
// ---------------------------------------------------------------------------
typedef __bf16 bf16_t;
typedef __attribute__((ext_vector_type(8)))  __bf16 v8bf;
typedef __attribute__((ext_vector_type(16))) __bf16 v16bf;
typedef __attribute__((ext_vector_type(8)))  float  v8f;

#define HIDDEN 1024
#define SEQ    2048
#define BATCH  2
#define HEADS  16
#define DK     64
#define MROWS  (BATCH * SEQ)   // 4096

// float -> bf16 round-to-nearest-even
__device__ __forceinline__ unsigned short f2bf(float f) {
  unsigned int u = __float_as_uint(f);
  unsigned int r = u + 0x7FFFu + ((u >> 16) & 1u);
  return (unsigned short)(r >> 16);
}

// Load a 16-bit WMMA fragment (A layout; the same bit pattern serves as the
// B-operand when the matrix is stored transposed, since B's lane-n column
// K-run mirrors A's lane-m row K-run).
//   lane 0-15 : row = lane,    K = 0+0..7   and 16..23
//   lane 16-31: row = lane-16, K = 8+0..7   and 24..31
__device__ __forceinline__ v16bf load_frag(const unsigned short* base, int ld) {
  int lane = (int)(threadIdx.x & 31u);
  int r    = lane & 15;
  int koff = (lane >> 4) << 3;
  const bf16_t* p = (const bf16_t*)base + r * ld + koff;
  v8bf lo = *(const v8bf*)(p);
  v8bf hi = *(const v8bf*)(p + 16);
  return __builtin_shufflevector(lo, hi, 0, 1, 2, 3, 4, 5, 6, 7,
                                 8, 9, 10, 11, 12, 13, 14, 15);
}

__device__ __forceinline__ v8f wmma_bf16(v16bf a, v16bf b, v8f c) {
  // (neg_a, A, neg_b, B, c_mod, C, reuse_a, reuse_b)
  return __builtin_amdgcn_wmma_f32_16x16x32_bf16(false, a, false, b,
                                                 (short)0, c, false, false);
}

// ---------------------------------------------------------------------------
// Weight convert+transpose: W (K x N, f32) -> Wt (N x K, bf16)
// ---------------------------------------------------------------------------
__global__ void transpose_bf16_kernel(const float* __restrict__ W,
                                      unsigned short* __restrict__ Wt,
                                      int K, int N) {
  int idx = blockIdx.x * 256 + threadIdx.x;
  if (idx >= K * N) return;
  int k = idx / N;
  int n = idx - k * N;            // coalesced read along n
  Wt[n * K + k] = f2bf(W[idx]);
}

// ---------------------------------------------------------------------------
// LayerNorm over HIDDEN, fp32 in -> bf16 out. One block (256 thr) per row.
// ---------------------------------------------------------------------------
__global__ __launch_bounds__(256)
void ln_kernel(const float* __restrict__ x, const float* __restrict__ g,
               const float* __restrict__ be, unsigned short* __restrict__ out) {
  __shared__ float s1[256], s2[256];
  int row = blockIdx.x;
  const float* xr = x + (size_t)row * HIDDEN;
  float v[4];
  float sum = 0.f, sq = 0.f;
#pragma unroll
  for (int i = 0; i < 4; i++) {
    v[i] = xr[threadIdx.x + i * 256];
    sum += v[i];
    sq  += v[i] * v[i];
  }
  s1[threadIdx.x] = sum;
  s2[threadIdx.x] = sq;
  __syncthreads();
  for (int off = 128; off > 0; off >>= 1) {
    if ((int)threadIdx.x < off) {
      s1[threadIdx.x] += s1[threadIdx.x + off];
      s2[threadIdx.x] += s2[threadIdx.x + off];
    }
    __syncthreads();
  }
  float mu  = s1[0] * (1.0f / HIDDEN);
  float var = s2[0] * (1.0f / HIDDEN) - mu * mu;
  float inv = rsqrtf(var + 1e-5f);
#pragma unroll
  for (int i = 0; i < 4; i++) {
    int idx = threadIdx.x + i * 256;
    float y = (v[i] - mu) * inv * g[idx] + be[idx];
    out[(size_t)row * HIDDEN + idx] = f2bf(y);
  }
}

// ---------------------------------------------------------------------------
// GEMM: C[M x N] = A(bf16, ld=Kd) @ Wt(bf16, N x Kd)^T + bias, fused epilogue.
// 256 threads = 8 waves in 2(M) x 4(N); each wave register-blocks a 32x32
// output (2 A-frags x 2 B-frags -> 4 WMMAs per 32-wide k-step), so every
// fragment load feeds two WMMAs. Block tile 64(M) x 128(N).
// mode 0: Q-proj  -> bf16 scatter to [b][h][s][d]
// mode 1: V-proj  -> bf16 scatter to [b][h][d][s]  (transposed for P@V)
// mode 2: f32 out = residual + acc + bias          (O-proj, FFN2)
// mode 3: bf16 out = gelu_tanh(acc + bias), row-major M x N (FFN1)
// ---------------------------------------------------------------------------
__global__ __launch_bounds__(256)
void gemm_kernel(const unsigned short* __restrict__ A,
                 const unsigned short* __restrict__ Wt,
                 const float* __restrict__ bias,
                 const float* __restrict__ residual,
                 unsigned short* __restrict__ out_bf,
                 float* __restrict__ out_f32,
                 int Kd, int N, int mode) {
  int wid  = (int)(threadIdx.x >> 5);
  int lane = (int)(threadIdx.x & 31u);
  int m0 = blockIdx.x * 64  + (wid & 1) * 32;   // wave's first M row
  int n0 = blockIdx.y * 128 + (wid >> 1) * 32;  // wave's first N col

  v8f z = {0.f, 0.f, 0.f, 0.f, 0.f, 0.f, 0.f, 0.f};
  v8f acc00 = z, acc01 = z, acc10 = z, acc11 = z;

  const unsigned short* Ab0 = A  + (size_t)m0 * Kd;
  const unsigned short* Ab1 = Ab0 + (size_t)16 * Kd;
  const unsigned short* Bb0 = Wt + (size_t)n0 * Kd;
  const unsigned short* Bb1 = Bb0 + (size_t)16 * Kd;

  for (int k = 0; k < Kd; k += 32) {
    v16bf a0 = load_frag(Ab0 + k, Kd);
    v16bf a1 = load_frag(Ab1 + k, Kd);
    v16bf b0 = load_frag(Bb0 + k, Kd);
    v16bf b1 = load_frag(Bb1 + k, Kd);
    acc00 = wmma_bf16(a0, b0, acc00);
    acc01 = wmma_bf16(a0, b1, acc01);
    acc10 = wmma_bf16(a1, b0, acc10);
    acc11 = wmma_bf16(a1, b1, acc11);
  }

  int g  = lane >> 4;
  int nc = lane & 15;
#pragma unroll
  for (int mi = 0; mi < 2; mi++) {
#pragma unroll
    for (int ni = 0; ni < 2; ni++) {
      v8f acc = (mi == 0) ? (ni == 0 ? acc00 : acc01)
                          : (ni == 0 ? acc10 : acc11);
      int col = n0 + ni * 16 + nc;
      float bcol = bias[col];
#pragma unroll
      for (int r = 0; r < 8; r++) {
        int row = m0 + mi * 16 + r + 8 * g;
        float val = acc[r] + bcol;
        if (mode == 0) {                    // Q: [b][h][s][d]
          int b_ = row >> 11, s = row & 2047, h = col >> 6, d = col & 63;
          out_bf[(((size_t)(b_ * HEADS + h) * SEQ) + s) * DK + d] = f2bf(val);
        } else if (mode == 1) {             // Vt: [b][h][d][s]
          int b_ = row >> 11, s = row & 2047, h = col >> 6, d = col & 63;
          out_bf[(((size_t)(b_ * HEADS + h) * DK) + d) * SEQ + s] = f2bf(val);
        } else if (mode == 2) {             // residual + proj
          size_t idx = (size_t)row * N + col;
          out_f32[idx] = residual[idx] + val;
        } else {                            // gelu(tanh)
          float xg = val;
          float t  = tanhf(0.7978845608028654f * (xg + 0.044715f * xg * xg * xg));
          out_bf[(size_t)row * N + col] = f2bf(0.5f * xg * (1.0f + t));
        }
      }
    }
  }
}

// ---------------------------------------------------------------------------
// Causal flash attention, Q==K (reference uses qk for both sides).
// One wave (32 thr) per 16-query tile per (b,h). Online softmax, P staged in
// LDS to convert C-fragment -> A-fragment layout, O over 4 dim-tiles.
// ---------------------------------------------------------------------------
__global__ __launch_bounds__(32)
void attn_kernel(const unsigned short* __restrict__ qk,
                 const unsigned short* __restrict__ vt,
                 unsigned short* __restrict__ aout) {
  __shared__ alignas(16) unsigned short pbuf[16 * 32];
  int lane = (int)(threadIdx.x & 31u);
  int qt = blockIdx.x & 127;           // 128 query tiles
  int bh = blockIdx.x >> 7;            // b*HEADS + h
  int b_ = bh >> 4, h = bh & 15;
  int q0 = qt * 16;

  const unsigned short* kq  = qk + (size_t)bh * SEQ * DK;
  const unsigned short* vtb = vt + (size_t)bh * DK * SEQ;

  v16bf aq0 = load_frag(kq + q0 * DK + 0, DK);
  v16bf aq1 = load_frag(kq + q0 * DK + 32, DK);

  v8f o0 = {0.f,0.f,0.f,0.f,0.f,0.f,0.f,0.f};
  v8f o1 = o0, o2 = o0, o3 = o0;
  float mrow[8], lrow[8];
#pragma unroll
  for (int r = 0; r < 8; r++) { mrow[r] = -1e30f; lrow[r] = 0.f; }

  int g  = lane >> 4;
  int nc = lane & 15;
  int kbmax = (q0 + 15) >> 5;

  for (int kb = 0; kb <= kbmax; kb++) {
    int kbase = kb * 32;
    v8f s0 = {0.f,0.f,0.f,0.f,0.f,0.f,0.f,0.f};
    v8f s1 = s0;
    // S = Q @ K^T over d=64: two 16-key halves x two k-steps
    s0 = wmma_bf16(aq0, load_frag(kq + (kbase +  0) * DK +  0, DK), s0);
    s0 = wmma_bf16(aq1, load_frag(kq + (kbase +  0) * DK + 32, DK), s0);
    s1 = wmma_bf16(aq0, load_frag(kq + (kbase + 16) * DK +  0, DK), s1);
    s1 = wmma_bf16(aq1, load_frag(kq + (kbase + 16) * DK + 32, DK), s1);

    float p0[8], p1[8];
#pragma unroll
    for (int r = 0; r < 8; r++) {
      int qrow = q0 + r + 8 * g;
      float a0 = (kbase + nc      <= qrow) ? s0[r] * 0.125f : -1e30f;
      float a1 = (kbase + 16 + nc <= qrow) ? s1[r] * 0.125f : -1e30f;
      float mx = fmaxf(a0, a1);
      mx = fmaxf(mx, __shfl_xor(mx, 1, 32));
      mx = fmaxf(mx, __shfl_xor(mx, 2, 32));
      mx = fmaxf(mx, __shfl_xor(mx, 4, 32));
      mx = fmaxf(mx, __shfl_xor(mx, 8, 32));
      float nm = fmaxf(mrow[r], mx);
      float sc = __expf(mrow[r] - nm);
      float e0 = __expf(a0 - nm);
      float e1 = __expf(a1 - nm);
      float rs = e0 + e1;
      rs += __shfl_xor(rs, 1, 32);
      rs += __shfl_xor(rs, 2, 32);
      rs += __shfl_xor(rs, 4, 32);
      rs += __shfl_xor(rs, 8, 32);
      lrow[r] = lrow[r] * sc + rs;
      mrow[r] = nm;
      o0[r] *= sc; o1[r] *= sc; o2[r] *= sc; o3[r] *= sc;
      p0[r] = e0; p1[r] = e1;
    }

    // C-fragment -> LDS -> A-fragment for P
    __syncthreads();
#pragma unroll
    for (int r = 0; r < 8; r++) {
      int Mr = r + 8 * g;
      pbuf[Mr * 32 + nc]      = f2bf(p0[r]);
      pbuf[Mr * 32 + 16 + nc] = f2bf(p1[r]);
    }
    __syncthreads();
    v16bf pa = load_frag(pbuf, 32);

    // O += P @ V over 32 keys, 4 dim tiles (Vt is [d][s] so B is contiguous)
    o0 = wmma_bf16(pa, load_frag(vtb +  0 * SEQ + kbase, SEQ), o0);
    o1 = wmma_bf16(pa, load_frag(vtb + 16 * SEQ + kbase, SEQ), o1);
    o2 = wmma_bf16(pa, load_frag(vtb + 32 * SEQ + kbase, SEQ), o2);
    o3 = wmma_bf16(pa, load_frag(vtb + 48 * SEQ + kbase, SEQ), o3);
    __syncthreads();
  }

#pragma unroll
  for (int r = 0; r < 8; r++) {
    float inv = 1.0f / lrow[r];
    int row = q0 + r + 8 * g;
    size_t base = ((size_t)(b_ * SEQ + row)) * HIDDEN + h * DK;
    aout[base +  0 + nc] = f2bf(o0[r] * inv);
    aout[base + 16 + nc] = f2bf(o1[r] * inv);
    aout[base + 32 + nc] = f2bf(o2[r] * inv);
    aout[base + 48 + nc] = f2bf(o3[r] * inv);
  }
}

// ---------------------------------------------------------------------------
// Host-side orchestration (graph-capture safe: launches only)
// ---------------------------------------------------------------------------
extern "C" void kernel_launch(void* const* d_in, const int* in_sizes, int n_in,
                              void* d_out, int out_size, void* d_ws, size_t ws_size,
                              hipStream_t stream) {
  const float* x     = (const float*)d_in[0];
  const float* Wq    = (const float*)d_in[1];
  const float* bq    = (const float*)d_in[2];
  const float* Wv    = (const float*)d_in[3];
  const float* bv    = (const float*)d_in[4];
  const float* Wo    = (const float*)d_in[5];
  const float* bo    = (const float*)d_in[6];
  const float* W1    = (const float*)d_in[7];
  const float* b1    = (const float*)d_in[8];
  const float* W2    = (const float*)d_in[9];
  const float* b2    = (const float*)d_in[10];
  const float* g1    = (const float*)d_in[11];
  const float* beta1 = (const float*)d_in[12];
  const float* g2    = (const float*)d_in[13];
  const float* beta2 = (const float*)d_in[14];

  char* ws = (char*)d_ws;
  const size_t MB = 1024u * 1024u;
  unsigned short* WqT = (unsigned short*)(ws +  0 * MB);  //  2 MB
  unsigned short* WvT = (unsigned short*)(ws +  2 * MB);  //  2 MB
  unsigned short* WoT = (unsigned short*)(ws +  4 * MB);  //  2 MB
  unsigned short* W1T = (unsigned short*)(ws +  6 * MB);  //  8 MB
  unsigned short* W2T = (unsigned short*)(ws + 14 * MB);  //  8 MB
  unsigned short* xn  = (unsigned short*)(ws + 22 * MB);  //  8 MB
  unsigned short* qb  = (unsigned short*)(ws + 30 * MB);  //  8 MB
  unsigned short* vtb = (unsigned short*)(ws + 38 * MB);  //  8 MB
  unsigned short* aob = (unsigned short*)(ws + 46 * MB);  //  8 MB
  float*          yb  = (float*)         (ws + 54 * MB);  // 16 MB
  unsigned short* hb  = (unsigned short*)(ws + 70 * MB);  //  8 MB
  unsigned short* ff1 = (unsigned short*)(ws + 78 * MB);  // 32 MB

  // Weight convert + transpose to bf16 N x K
  transpose_bf16_kernel<<<(HIDDEN * HIDDEN + 255) / 256, 256, 0, stream>>>(Wq, WqT, HIDDEN, HIDDEN);
  transpose_bf16_kernel<<<(HIDDEN * HIDDEN + 255) / 256, 256, 0, stream>>>(Wv, WvT, HIDDEN, HIDDEN);
  transpose_bf16_kernel<<<(HIDDEN * HIDDEN + 255) / 256, 256, 0, stream>>>(Wo, WoT, HIDDEN, HIDDEN);
  transpose_bf16_kernel<<<(HIDDEN * 4 * HIDDEN + 255) / 256, 256, 0, stream>>>(W1, W1T, HIDDEN, 4 * HIDDEN);
  transpose_bf16_kernel<<<(4 * HIDDEN * HIDDEN + 255) / 256, 256, 0, stream>>>(W2, W2T, 4 * HIDDEN, HIDDEN);

  // LN1
  ln_kernel<<<MROWS, 256, 0, stream>>>(x, g1, beta1, xn);

  // Q/V projections (Q doubles as K in the reference)
  dim3 gproj(MROWS / 64, HIDDEN / 128);
  gemm_kernel<<<gproj, 256, 0, stream>>>(xn, WqT, bq, nullptr, qb, nullptr, HIDDEN, HIDDEN, 0);
  gemm_kernel<<<gproj, 256, 0, stream>>>(xn, WvT, bv, nullptr, vtb, nullptr, HIDDEN, HIDDEN, 1);

  // Causal attention
  attn_kernel<<<BATCH * HEADS * (SEQ / 16), 32, 0, stream>>>(qb, vtb, aob);

  // O-projection + residual -> yb (f32)
  gemm_kernel<<<gproj, 256, 0, stream>>>(aob, WoT, bo, x, nullptr, yb, HIDDEN, HIDDEN, 2);

  // LN2
  ln_kernel<<<MROWS, 256, 0, stream>>>(yb, g2, beta2, hb);

  // FFN1 with GELU -> bf16
  dim3 gff1(MROWS / 64, (4 * HIDDEN) / 128);
  gemm_kernel<<<gff1, 256, 0, stream>>>(hb, W1T, b1, nullptr, ff1, nullptr, HIDDEN, 4 * HIDDEN, 3);

  // FFN2 + residual -> d_out (f32)
  gemm_kernel<<<gproj, 256, 0, stream>>>(ff1, W2T, b2, yb, nullptr, (float*)d_out, 4 * HIDDEN, HIDDEN, 2);
}
